// graph_rcnn_84610855731242
// MI455X (gfx1250) — matrix operand order repaired and checked by reading, not compiled, not run on hardware
//
#include <hip/hip_runtime.h>
#include <hip/hip_bf16.h>

// ---------------- CDNA5 WMMA types ----------------
typedef __attribute__((ext_vector_type(16))) __bf16 v16bf;
typedef __attribute__((ext_vector_type(8)))  __bf16 v8bf;
typedef __attribute__((ext_vector_type(8)))  float  v8f;

#define DEMB 512
#define DNODE 2048
#define DEDGE 1024
#define NB 64
#define PB 128
#define BB 8

// Combine two 16-byte LDS chunks into one v16bf fragment register set.
__device__ __forceinline__ v16bf mk16(const __bf16* p0, const __bf16* p1) {
    v8bf lo = *(const v8bf*)p0;
    v8bf hi = *(const v8bf*)p1;
    v16bf r;
#pragma unroll
    for (int i = 0; i < 8; ++i) { r[i] = lo[i]; r[i + 8] = hi[i]; }
    return r;
}

// A fragment (16x32 bf16, ISA 7.12.2): lanes 0-15 row=lane, K {0..7,16..23};
// lanes 16-31 row=lane-16, K {8..15,24..31}. base = &tile[row][0], ld in elems.
__device__ __forceinline__ v16bf frag_a(const __bf16* rowp, int half) {
    return mk16(rowp + half * 8, rowp + 16 + half * 8);
}
// B fragment (32x16 bf16): lane holds one column's 16 K values,
// lanes 0-15 -> K {0..7,16..23}, lanes 16-31 -> K {8..15,24..31}. rowp = &tileT[col][0].
__device__ __forceinline__ v16bf frag_b(const __bf16* rowp, int half) {
    return mk16(rowp + half * 8, rowp + 16 + half * 8);
}

__device__ __forceinline__ void cvt8_store(__bf16* dst, float4 v0, float4 v1) {
    dst[0] = (__bf16)v0.x; dst[1] = (__bf16)v0.y; dst[2] = (__bf16)v0.z; dst[3] = (__bf16)v0.w;
    dst[4] = (__bf16)v1.x; dst[5] = (__bf16)v1.y; dst[6] = (__bf16)v1.z; dst[7] = (__bf16)v1.w;
}

// ============ Kernel A: node projection  out[M=512,512] = X[512,K] @ W[K,512] + b ============
__global__ __launch_bounds__(256)
void proj_kernel(const float* __restrict__ X, const float* __restrict__ W,
                 const float* __restrict__ bias, float* __restrict__ out, int K) {
    __shared__ __align__(16) __bf16 As[64][40];   // [m][k]
    __shared__ __align__(16) __bf16 Bs[64][40];   // [n][k] (transposed W tile)
    const int tid = threadIdx.x, lane = tid & 31, wid = tid >> 5;
    const int mBase = blockIdx.x * 64, nBase = blockIdx.y * 64;
    const int waveM = (wid & 3) * 16, waveN = (wid >> 2) * 32;
    const int half = lane >> 4, lr = lane & 15;
    const int arow = tid >> 2, akk = (tid & 3) * 8;   // A: 64 rows x 32 k
    const int bk = tid >> 3, bno = (tid & 7) * 8;     // B: 32 k-rows x 64 n

    v8f acc0 = {}, acc1 = {};
    for (int k0 = 0; k0 < K; k0 += 32) {
        __syncthreads();
        {   // stage A (fp32 -> bf16)
            const float4* s = (const float4*)(X + (size_t)(mBase + arow) * K + k0 + akk);
            cvt8_store(&As[arow][akk], s[0], s[1]);
        }
        {   // stage W transposed
            const float4* s = (const float4*)(W + (size_t)(k0 + bk) * DEMB + nBase + bno);
            float4 v0 = s[0], v1 = s[1];
            Bs[bno + 0][bk] = (__bf16)v0.x; Bs[bno + 1][bk] = (__bf16)v0.y;
            Bs[bno + 2][bk] = (__bf16)v0.z; Bs[bno + 3][bk] = (__bf16)v0.w;
            Bs[bno + 4][bk] = (__bf16)v1.x; Bs[bno + 5][bk] = (__bf16)v1.y;
            Bs[bno + 6][bk] = (__bf16)v1.z; Bs[bno + 7][bk] = (__bf16)v1.w;
        }
        if (k0 + 32 < K)  // global_prefetch_b8 of next A chunk
            __builtin_prefetch(X + (size_t)(mBase + arow) * K + k0 + 32 + akk, 0, 1);
        __syncthreads();
        v16bf a  = frag_a(&As[waveM + lr][0], half);
        v16bf b0 = frag_b(&Bs[waveN + lr][0], half);
        v16bf b1 = frag_b(&Bs[waveN + 16 + lr][0], half);
        acc0 = __builtin_amdgcn_wmma_f32_16x16x32_bf16(false, a, false, b0, (short)0, acc0, false, false);
        acc1 = __builtin_amdgcn_wmma_f32_16x16x32_bf16(false, a, false, b1, (short)0, acc1, false, false);
    }
    const int rowOff = half * 8;
#pragma unroll
    for (int r = 0; r < 8; ++r) {
        int grow = mBase + waveM + r + rowOff;
        int gc0 = nBase + waveN + lr, gc1 = gc0 + 16;
        out[(size_t)grow * DEMB + gc0] = acc0[r] + bias[gc0];
        out[(size_t)grow * DEMB + gc1] = acc1[r] + bias[gc1];
    }
}

// ============ Kernel B: relatedness = sigmoid(phi[b] @ psi[b]^T), 64x64 per batch ============
__global__ __launch_bounds__(256)
void score_kernel(const float* __restrict__ phi, const float* __restrict__ psi,
                  float* __restrict__ out) {
    __shared__ __align__(16) __bf16 As[64][40];   // phi rows [i][k]
    __shared__ __align__(16) __bf16 Bs[64][40];   // psi rows [j][k] (already B-friendly)
    const int b = blockIdx.x;
    const int tid = threadIdx.x, lane = tid & 31, wid = tid >> 5;
    const int waveM = (wid & 3) * 16, waveN = (wid >> 2) * 32;
    const int half = lane >> 4, lr = lane & 15;
    const int arow = tid >> 2, akk = (tid & 3) * 8;

    v8f acc0 = {}, acc1 = {};
    const size_t base = (size_t)b * 64 * DEMB;
    for (int k0 = 0; k0 < DEMB; k0 += 32) {
        __syncthreads();
        {
            const float4* s = (const float4*)(phi + base + (size_t)arow * DEMB + k0 + akk);
            cvt8_store(&As[arow][akk], s[0], s[1]);
        }
        {
            const float4* s = (const float4*)(psi + base + (size_t)arow * DEMB + k0 + akk);
            cvt8_store(&Bs[arow][akk], s[0], s[1]);
        }
        __syncthreads();
        v16bf a  = frag_a(&As[waveM + lr][0], half);
        v16bf b0 = frag_b(&Bs[waveN + lr][0], half);
        v16bf b1 = frag_b(&Bs[waveN + 16 + lr][0], half);
        acc0 = __builtin_amdgcn_wmma_f32_16x16x32_bf16(false, a, false, b0, (short)0, acc0, false, false);
        acc1 = __builtin_amdgcn_wmma_f32_16x16x32_bf16(false, a, false, b1, (short)0, acc1, false, false);
    }
    const int rowOff = half * 8;
#pragma unroll
    for (int r = 0; r < 8; ++r) {
        int i = waveM + r + rowOff;
        int j0 = waveN + lr, j1 = j0 + 16;
        out[(size_t)b * 4096 + i * 64 + j0] = 1.0f / (1.0f + __expf(-acc0[r]));
        out[(size_t)b * 4096 + i * 64 + j1] = 1.0f / (1.0f + __expf(-acc1[r]));
    }
}

// ============ Kernel C: second_half = 0.5*(IF[b,i0,i1]+IF[b,i1,i0]) @ W_rel + b_rel ============
__global__ __launch_bounds__(256)
void edge_kernel(const float* __restrict__ IF, const float* __restrict__ Wrel,
                 const float* __restrict__ brel, const int* __restrict__ pairs,
                 float* __restrict__ out) {
    __shared__ __align__(16) __bf16 As[64][40];
    __shared__ __align__(16) __bf16 Bs[64][40];
    __shared__ int offF[64], offR[64];
    const int tid = threadIdx.x, lane = tid & 31, wid = tid >> 5;
    const int mBase = blockIdx.x * 64, nBase = blockIdx.y * 64;
    const int waveM = (wid & 3) * 16, waveN = (wid >> 2) * 32;
    const int half = lane >> 4, lr = lane & 15;
    const int arow = tid >> 2, akk = (tid & 3) * 8;
    const int bk = tid >> 3, bno = (tid & 7) * 8;

    if (tid < 64) {
        int gr = mBase + tid;              // global pair-row 0..1023
        int b = gr >> 7, p = gr & 127;
        int i0 = pairs[(b * PB + p) * 2 + 0];
        int i1 = pairs[(b * PB + p) * 2 + 1];
        // edge lookup collapses to index arithmetic (elt is the identity meshgrid)
        offF[tid] = ((b * NB + i0) * NB + i1) * DEDGE;
        offR[tid] = ((b * NB + i1) * NB + i0) * DEDGE;
    }
    v8f acc0 = {}, acc1 = {};
    for (int k0 = 0; k0 < DEDGE; k0 += 32) {
        __syncthreads();
        {   // gather fwd+rev rows, average, convert
            const float4* f = (const float4*)(IF + (size_t)offF[arow] + k0 + akk);
            const float4* g = (const float4*)(IF + (size_t)offR[arow] + k0 + akk);
            float4 f0 = f[0], f1 = f[1], g0 = g[0], g1 = g[1];
            float4 v0, v1;
            v0.x = 0.5f * (f0.x + g0.x); v0.y = 0.5f * (f0.y + g0.y);
            v0.z = 0.5f * (f0.z + g0.z); v0.w = 0.5f * (f0.w + g0.w);
            v1.x = 0.5f * (f1.x + g1.x); v1.y = 0.5f * (f1.y + g1.y);
            v1.z = 0.5f * (f1.z + g1.z); v1.w = 0.5f * (f1.w + g1.w);
            cvt8_store(&As[arow][akk], v0, v1);
        }
        {   // stage W_rel transposed
            const float4* s = (const float4*)(Wrel + (size_t)(k0 + bk) * DEMB + nBase + bno);
            float4 v0 = s[0], v1 = s[1];
            Bs[bno + 0][bk] = (__bf16)v0.x; Bs[bno + 1][bk] = (__bf16)v0.y;
            Bs[bno + 2][bk] = (__bf16)v0.z; Bs[bno + 3][bk] = (__bf16)v0.w;
            Bs[bno + 4][bk] = (__bf16)v1.x; Bs[bno + 5][bk] = (__bf16)v1.y;
            Bs[bno + 6][bk] = (__bf16)v1.z; Bs[bno + 7][bk] = (__bf16)v1.w;
        }
        __syncthreads();
        v16bf a  = frag_a(&As[waveM + lr][0], half);
        v16bf b0 = frag_b(&Bs[waveN + lr][0], half);
        v16bf b1 = frag_b(&Bs[waveN + 16 + lr][0], half);
        acc0 = __builtin_amdgcn_wmma_f32_16x16x32_bf16(false, a, false, b0, (short)0, acc0, false, false);
        acc1 = __builtin_amdgcn_wmma_f32_16x16x32_bf16(false, a, false, b1, (short)0, acc1, false, false);
    }
    const int rowOff = half * 8;
    float* cls = out + BB * NB * NB;   // classifier_input base
#pragma unroll
    for (int r = 0; r < 8; ++r) {
        int grow = mBase + waveM + r + rowOff;
        int gc0 = nBase + waveN + lr, gc1 = gc0 + 16;
        cls[(size_t)grow * 1024 + 512 + gc0] = acc0[r] + brel[gc0];
        cls[(size_t)grow * 1024 + 512 + gc1] = acc1[r] + brel[gc1];
    }
}

// ============ Kernel D: first_half = 0.5*(node_emb[b,i0] + node_emb[b,i1]) ============
__global__ __launch_bounds__(128)
void first_half_kernel(const float* __restrict__ ne, const int* __restrict__ pairs,
                       float* __restrict__ out) {
    const int row = blockIdx.x;             // 0..1023
    const int b = row >> 7, p = row & 127;
    const int i0 = pairs[(b * PB + p) * 2 + 0];
    const int i1 = pairs[(b * PB + p) * 2 + 1];
    const float4* a = (const float4*)(ne + (size_t)(b * NB + i0) * DEMB);
    const float4* c = (const float4*)(ne + (size_t)(b * NB + i1) * DEMB);
    float4* o = (float4*)(out + BB * NB * NB + (size_t)row * 1024);
    const int t = threadIdx.x;              // 128 threads * float4 = 512 cols
    float4 va = a[t], vc = c[t], r;
    r.x = 0.5f * (va.x + vc.x); r.y = 0.5f * (va.y + vc.y);
    r.z = 0.5f * (va.z + vc.z); r.w = 0.5f * (va.w + vc.w);
    o[t] = r;
}

extern "C" void kernel_launch(void* const* d_in, const int* in_sizes, int n_in,
                              void* d_out, int out_size, void* d_ws, size_t ws_size,
                              hipStream_t stream) {
    (void)in_sizes; (void)n_in; (void)out_size; (void)ws_size;
    const float* x      = (const float*)d_in[0];
    const float* IF     = (const float*)d_in[1];
    const float* W_phi  = (const float*)d_in[2];
    const float* b_phi  = (const float*)d_in[3];
    const float* W_psi  = (const float*)d_in[4];
    const float* b_psi  = (const float*)d_in[5];
    const float* W_node = (const float*)d_in[6];
    const float* b_node = (const float*)d_in[7];
    const float* W_rel  = (const float*)d_in[8];
    const float* b_rel  = (const float*)d_in[9];
    const int*   pairs  = (const int*)d_in[10];
    float* out = (float*)d_out;

    float* phi = (float*)d_ws;              // 512*512 f32
    float* psi = phi + 512 * 512;
    float* ne  = psi + 512 * 512;

    dim3 blk(256);
    dim3 gP(8, 8);                          // 512/64 x 512/64
    proj_kernel<<<gP, blk, 0, stream>>>(x, W_phi,  b_phi,  phi, DNODE);
    proj_kernel<<<gP, blk, 0, stream>>>(x, W_psi,  b_psi,  psi, DNODE);
    proj_kernel<<<gP, blk, 0, stream>>>(x, W_node, b_node, ne,  DNODE);
    score_kernel<<<BB, blk, 0, stream>>>(phi, psi, out);
    edge_kernel<<<dim3(16, 8), blk, 0, stream>>>(IF, W_rel, b_rel, pairs, out);
    first_half_kernel<<<1024, 128, 0, stream>>>(ne, pairs, out);
}
